// RRFU_76295799046325
// MI455X (gfx1250) — matrix operand order repaired and checked by reading, not compiled, not run on hardware
//
#include <hip/hip_runtime.h>
#include <hip/hip_bf16.h>

// Problem constants (from the reference)
#define SS    2
#define GG    4
#define KK    3
#define CC    64
#define BB    4
#define HH    128
#define WW    128
#define HOo   256          // S*H
#define WOo   256          // S*W
#define CGRP  16           // C/G
#define OCOFF 32           // G*S*S*2
#define KC9   9            // K*K

// Padded x_def plane (zero ring for branch-free 3x3 gather)
#define PWO (WOo + 2)      // 258
#define PHO (HOo + 2)      // 258

// Kernel-3 tile
#define TW  32
#define TH  8
#define TWP (TW + 2)       // 34
#define THP (TH + 2)       // 10

typedef __attribute__((ext_vector_type(16))) _Float16 v16h;
typedef __attribute__((ext_vector_type(8)))  float    v8f;
typedef __attribute__((ext_vector_type(4)))  unsigned int u32x4;
typedef __attribute__((ext_vector_type(8)))  int      i32x8;
typedef __attribute__((ext_vector_type(4)))  int      i32x4;

__device__ __forceinline__ float sigmoidf_(float v) {
  return 1.0f / (1.0f + __expf(-v));
}

__device__ __forceinline__ v8f wmma_f16(v16h a, v16h b, v8f c) {
  // D(16x16,f32) = A(16x32,f16) * B(32x16,f16) + C
  return __builtin_amdgcn_wmma_f32_16x16x32_f16(false, a, false, b, (short)0, c,
                                                false, false);
}

// A fragment (16x32 f16, MxK) from f32 row-major weights w[rows][64].
// lane<16: elems 0..15 = K(kBase+0..15)
// lane>=16: elems 0..15 = K(kBase+16..31)
__device__ __forceinline__ v16h load_A_frag(const float* __restrict__ w, int rows,
                                            int mBase, int kBase, int lane) {
  int m  = mBase + (lane & 15);
  int kh = (lane < 16) ? 0 : 16;
  v16h a;
  if (m < rows) {
    const float* wr = w + (size_t)m * CC;
#pragma unroll
    for (int j = 0; j < 8; ++j) a[j]     = (_Float16)wr[kBase + kh + j];
#pragma unroll
    for (int j = 0; j < 8; ++j) a[8 + j] = (_Float16)wr[kBase + kh + 8 + j];
  } else {
#pragma unroll
    for (int j = 0; j < 16; ++j) a[j] = (_Float16)0.0f;
  }
  return a;
}

// B fragment (32x16 f16, KxN) from channel-planar f32 global memory.
__device__ __forceinline__ v16h load_B_frag(const float* __restrict__ xpix, int kBase,
                                            size_t cstride, int lane) {
  int c0 = kBase + ((lane < 16) ? 0 : 16);
  int n  = lane & 15;
  v16h b;
#pragma unroll
  for (int j = 0; j < 16; ++j) b[j] = (_Float16)xpix[(size_t)(c0 + j) * cstride + n];
  return b;
}

// B fragment from the LDS halo tile: pixel n at tile[c][ty][tx0+n].
__device__ __forceinline__ v16h load_B_lds(const float tile[CC][THP][TWP], int kBase,
                                           int ty, int tx0, int lane) {
  int c0 = kBase + ((lane < 16) ? 0 : 16);
  int n  = lane & 15;
  v16h b;
#pragma unroll
  for (int j = 0; j < 16; ++j) b[j] = (_Float16)tile[c0 + j][ty][tx0 + n];
  return b;
}

// ---------------------------------------------------------------------------
// Kernel 0: zero the 1-pixel ring of the padded x_def planes (B*C planes).
// ---------------------------------------------------------------------------
__global__ __launch_bounds__(256) void k_zero_ring(float* __restrict__ x_def_pad) {
  const int per   = 2 * PWO + 2 * (PHO - 2);   // 1028 ring elements per plane
  const int total = BB * CC * per;
  int idx = blockIdx.x * blockDim.x + threadIdx.x;
  if (idx >= total) return;
  const int plane_i = idx / per;
  const int r       = idx % per;
  float* p = x_def_pad + (size_t)plane_i * PHO * PWO;
  if (r < PWO) {
    p[r] = 0.0f;                                            // top row
  } else if (r < 2 * PWO) {
    p[(size_t)(PHO - 1) * PWO + (r - PWO)] = 0.0f;          // bottom row
  } else {
    const int rr  = r - 2 * PWO;
    const int row = 1 + (rr >> 1);
    const int col = (rr & 1) ? (PWO - 1) : 0;
    p[(size_t)row * PWO + col] = 0.0f;                      // side columns
  }
}

// ---------------------------------------------------------------------------
// Kernel 1: gated offset convs.  off_full[b, o, h, w], o in 0..31.
// One wave = 16 consecutive low-res pixels; GEMM M=32, N=16, K=64 via 8 WMMAs.
// ---------------------------------------------------------------------------
__global__ __launch_bounds__(256) void k_offsets(
    const float* __restrict__ x,
    const float* __restrict__ w_off,  const float* __restrict__ b_off,
    const float* __restrict__ w_offm, const float* __restrict__ b_offm,
    float* __restrict__ off_full) {
  const int lane  = threadIdx.x & 31;
  const int gwave = blockIdx.x * (blockDim.x >> 5) + (threadIdx.x >> 5);
  const int pix0  = gwave * 16;                       // linear over B*H*W (row-aligned)
  const int b  = pix0 / (HH * WW);
  const int r0 = pix0 % (HH * WW);
  const int h  = r0 / WW;
  const int w0 = r0 % WW;
  const size_t plane = (size_t)HH * WW;
  const float* xb = x + (size_t)b * CC * plane + (size_t)h * WW + w0;

  v16h B0 = load_B_frag(xb, 0,  plane, lane);
  v16h B1 = load_B_frag(xb, 32, plane, lane);

  v16h A00 = load_A_frag(w_off,  OCOFF, 0,  0,  lane);
  v16h A01 = load_A_frag(w_off,  OCOFF, 0,  32, lane);
  v16h A10 = load_A_frag(w_off,  OCOFF, 16, 0,  lane);
  v16h A11 = load_A_frag(w_off,  OCOFF, 16, 32, lane);
  v16h M00 = load_A_frag(w_offm, OCOFF, 0,  0,  lane);
  v16h M01 = load_A_frag(w_offm, OCOFF, 0,  32, lane);
  v16h M10 = load_A_frag(w_offm, OCOFF, 16, 0,  lane);
  v16h M11 = load_A_frag(w_offm, OCOFF, 16, 32, lane);

  v8f o0 = {}; o0 = wmma_f16(A00, B0, o0); o0 = wmma_f16(A01, B1, o0);
  v8f o1 = {}; o1 = wmma_f16(A10, B0, o1); o1 = wmma_f16(A11, B1, o1);
  v8f g0 = {}; g0 = wmma_f16(M00, B0, g0); g0 = wmma_f16(M01, B1, g0);
  v8f g1 = {}; g1 = wmma_f16(M10, B0, g1); g1 = wmma_f16(M11, B1, g1);

  const int mOff = (lane >> 4) * 8;   // C/D layout: VGPR r -> M = mOff + r, N = lane & 15
  const int n    = lane & 15;
  float* ob = off_full + (size_t)b * OCOFF * plane + (size_t)h * WW + w0 + n;
#pragma unroll
  for (int r = 0; r < 8; ++r) {
    int m = mOff + r;
    ob[(size_t)m * plane]        = (o0[r] + b_off[m])      * sigmoidf_(g0[r] + b_offm[m]);
    ob[(size_t)(m + 16) * plane] = (o1[r] + b_off[m + 16]) * sigmoidf_(g1[r] + b_offm[m + 16]);
  }
}

// ---------------------------------------------------------------------------
// Kernel 2: deformable bilinear upsample (border clamp) -> padded x_def.
// One thread per (bg, ho, wo), loops the group's 16 channels.
// ---------------------------------------------------------------------------
__global__ __launch_bounds__(256) void k_sample(
    const float* __restrict__ x, const float* __restrict__ off_full,
    float* __restrict__ x_def_pad) {
  const size_t total = (size_t)BB * GG * HOo * WOo;
  size_t idx = (size_t)blockIdx.x * blockDim.x + threadIdx.x;
  if (idx >= total) return;
  const int wo = (int)(idx % WOo);
  size_t t = idx / WOo;
  const int ho = (int)(t % HOo);
  const int bg = (int)(t / HOo);
  const int b = bg / GG, g = bg % GG;
  const int h0 = ho >> 1, w0 = wo >> 1, sy = ho & 1, sx = wo & 1;
  const int och = ((g * 2 + sy) * 2 + sx) * 2;       // channel of off_x; off_y = och+1

  const size_t plane = (size_t)HH * WW;
  const float* ob = off_full + (size_t)b * OCOFF * plane + (size_t)h0 * WW + w0;
  const float offx = ob[(size_t)och * plane];
  const float offy = ob[(size_t)(och + 1) * plane];

  float ix = ((float)wo + 0.5f + offx) * 0.5f - 0.5f;
  float iy = ((float)ho + 0.5f + offy) * 0.5f - 0.5f;
  ix = fminf(fmaxf(ix, 0.0f), (float)(WW - 1));
  iy = fminf(fmaxf(iy, 0.0f), (float)(HH - 1));
  const float fx = floorf(ix), fy = floorf(iy);
  const float wx = ix - fx,   wy = iy - fy;
  const int x0 = (int)fx, y0 = (int)fy;
  const int x1 = min(x0 + 1, WW - 1), y1 = min(y0 + 1, HH - 1);
  const float w00 = (1.0f - wx) * (1.0f - wy), w01 = wx * (1.0f - wy);
  const float w10 = (1.0f - wx) * wy,          w11 = wx * wy;

  const float* xb = x + ((size_t)b * CC + (size_t)g * CGRP) * plane;
  // cover gather latency for the first channels (global_prefetch_b8)
  __builtin_prefetch(xb + (size_t)y0 * WW + x0, 0, 0);
  __builtin_prefetch(xb + (size_t)y1 * WW + x0, 0, 0);

  const size_t pplane = (size_t)PHO * PWO;
  float* xd = x_def_pad + ((size_t)b * CC + (size_t)g * CGRP) * pplane +
              (size_t)(ho + 1) * PWO + (wo + 1);
#pragma unroll
  for (int cg = 0; cg < CGRP; ++cg) {
    const float* xc = xb + (size_t)cg * plane;
    float v = xc[y0 * WW + x0] * w00 + xc[y0 * WW + x1] * w01 +
              xc[y1 * WW + x0] * w10 + xc[y1 * WW + x1] * w11;
    xd[(size_t)cg * pplane] = v;
  }
}

// ---------------------------------------------------------------------------
// Kernel 3: TDM-staged tile + gated 9-ch convs (WMMA) + softmax + branch-free
// 3x3 weighted gather from LDS.  Block = 32x8 output pixels, 8 waves; each
// wave owns one row and two 16-pixel segments.
// ---------------------------------------------------------------------------
__global__ __launch_bounds__(256) void k_carafe(
    const float* __restrict__ x_def_pad,
    const float* __restrict__ w_sk, const float* __restrict__ b_sk,
    const float* __restrict__ w_km, const float* __restrict__ b_km,
    float* __restrict__ out) {
  __shared__ float tile[CC][THP][TWP];     // 64 x 10 x 34 f32 = 87,040 B
  __shared__ float stage[8][16][17];       // per-wave softmax staging
  const int lane = threadIdx.x & 31;
  const int wib  = threadIdx.x >> 5;

  const int nbx = WOo / TW;                // 8
  const int nby = HOo / TH;                // 32
  const int bx  = blockIdx.x % nbx;
  const int by  = (blockIdx.x / nbx) % nby;
  const int b   = blockIdx.x / (nbx * nby);
  const int wo0 = bx * TW;
  const int ho0 = by * TH;
  const size_t pplane = (size_t)PHO * PWO;
  const float* src = x_def_pad + (size_t)b * CC * pplane + (size_t)ho0 * PWO + wo0;

  // --- Tensor Data Mover: DMA (c=64, y=10, x=34) halo tile into LDS ---
  if (threadIdx.x < 32) {
    const unsigned ldsb = (unsigned)(size_t)(&tile[0][0][0]);
    const unsigned long long ga = (unsigned long long)(size_t)src;
    u32x4 g0 = {};
    g0.x = 1u;                                                   // count=1, user D#
    g0.y = ldsb;                                                 // lds_addr
    g0.z = (unsigned)ga;                                         // global_addr[31:0]
    g0.w = (unsigned)((ga >> 32) & 0x01FFFFFFull) | (2u << 30);  // addr[56:32] | type=2
    i32x8 g1 = {};
    g1[0] = (int)(2u << 16);                                     // data_size = 4 bytes
    g1[1] = (int)((unsigned)(PWO & 0xFFFF) << 16);               // tensor_dim0[15:0]
    g1[2] = (int)((unsigned)(PHO & 0xFFFF) << 16);               // tensor_dim1[15:0]
    g1[3] = (int)((unsigned)TWP << 16);                          // tile_dim0 = 34
    g1[4] = (int)((unsigned)THP | ((unsigned)CC << 16));         // tile_dim1=10, tile_dim2=64
    g1[5] = (int)PWO;                                            // tensor_dim0_stride = 258
    g1[6] = (int)(((unsigned)pplane & 0xFFFFu) << 16);           // tensor_dim1_stride[15:0]
    g1[7] = (int)((unsigned)(pplane >> 16));                     // tensor_dim1_stride[47:16]
    i32x4 g2 = {};
    g2[0] = CC;                                                  // tensor_dim2 = 64
    i32x4 g3 = {};
#if defined(__clang_major__) && (__clang_major__ >= 23)
    i32x8 g4 = {};
    __builtin_amdgcn_tensor_load_to_lds(g0, g1, g2, g3, g4, 0);
#else
    __builtin_amdgcn_tensor_load_to_lds(g0, g1, g2, g3, 0);
#endif
    __builtin_amdgcn_s_wait_tensorcnt(0);
  }
  __syncthreads();

  // A fragments (loop invariant), M padded 9 -> 16 with zero rows
  v16h S0 = load_A_frag(w_sk, KC9, 0, 0,  lane);
  v16h S1 = load_A_frag(w_sk, KC9, 0, 32, lane);
  v16h K0 = load_A_frag(w_km, KC9, 0, 0,  lane);
  v16h K1 = load_A_frag(w_km, KC9, 0, 32, lane);

  const int mOff = (lane >> 4) * 8;
  const int n    = lane & 15;

  for (int s = 0; s < 2; ++s) {
    const int tx0 = s * 16 + 1;        // tile col of first output pixel
    const int ty  = wib + 1;           // tile row of this wave's output row

    v16h B0 = load_B_lds(tile, 0,  ty, tx0, lane);
    v16h B1 = load_B_lds(tile, 32, ty, tx0, lane);

    v8f as = {}; as = wmma_f16(S0, B0, as); as = wmma_f16(S1, B1, as);
    v8f ak = {}; ak = wmma_f16(K0, B0, ak); ak = wmma_f16(K1, B1, ak);

#pragma unroll
    for (int r = 0; r < 8; ++r) {
      int m = mOff + r;
      float sv = 0.0f;
      if (m < KC9)
        sv = (as[r] + b_sk[m]) * sigmoidf_(ak[r] + b_km[m]);
      stage[wib][n][m] = sv;
    }
    __syncthreads();

    // softmax over the 9 kernel logits, one pixel per lane (lanes 0..15)
    if (lane < 16) {
      float mx = -3.402823466e38f;
#pragma unroll
      for (int j = 0; j < KC9; ++j) mx = fmaxf(mx, stage[wib][lane][j]);
      float e[KC9];
      float sum = 0.0f;
#pragma unroll
      for (int j = 0; j < KC9; ++j) { e[j] = __expf(stage[wib][lane][j] - mx); sum += e[j]; }
      const float inv = 1.0f / sum;
#pragma unroll
      for (int j = 0; j < KC9; ++j) stage[wib][lane][j] = e[j] * inv;
    }
    __syncthreads();

    // branch-free 3x3 weighted gather from the LDS tile (zero ring handles edges)
    for (int t0 = lane; t0 < 16 * CC; t0 += 32) {
      const int p = t0 & 15;
      const int c = t0 >> 4;
      float acc = 0.0f;
#pragma unroll
      for (int ky = 0; ky < KK; ++ky)
#pragma unroll
        for (int kx = 0; kx < KK; ++kx)
          acc += stage[wib][p][ky * KK + kx] * tile[c][wib + ky][s * 16 + p + kx];
      out[((size_t)b * CC + c) * ((size_t)HOo * WOo) +
          (size_t)(ho0 + wib) * WOo + (wo0 + s * 16 + p)] = acc;
    }
    __syncthreads();
  }
}

// ---------------------------------------------------------------------------
extern "C" void kernel_launch(void* const* d_in, const int* in_sizes, int n_in,
                              void* d_out, int out_size, void* d_ws, size_t ws_size,
                              hipStream_t stream) {
  (void)in_sizes; (void)n_in; (void)out_size; (void)ws_size;
  const float* x      = (const float*)d_in[0];
  const float* w_off  = (const float*)d_in[1];
  const float* b_off  = (const float*)d_in[2];
  const float* w_offm = (const float*)d_in[3];
  const float* b_offm = (const float*)d_in[4];
  const float* w_sk   = (const float*)d_in[5];
  const float* b_sk   = (const float*)d_in[6];
  const float* w_km   = (const float*)d_in[7];
  const float* b_km   = (const float*)d_in[8];
  float* out = (float*)d_out;

  // workspace: off_full (8 MB) then padded x_def (68 MB)
  float* off_full  = (float*)d_ws;
  float* x_def_pad = off_full + (size_t)BB * OCOFF * HH * WW;

  {
    const int total  = BB * CC * (2 * PWO + 2 * (PHO - 2));
    const int blocks = (total + 255) / 256;
    k_zero_ring<<<blocks, 256, 0, stream>>>(x_def_pad);
  }
  {
    const int waves  = (BB * HH * WW) / 16;   // 4096
    const int blocks = waves / 8;             // 512 blocks of 8 waves
    k_offsets<<<blocks, 256, 0, stream>>>(x, w_off, b_off, w_offm, b_offm, off_full);
  }
  {
    const size_t total = (size_t)BB * GG * HOo * WOo;   // 1,048,576
    const int blocks = (int)((total + 255) / 256);
    k_sample<<<blocks, 256, 0, stream>>>(x, off_full, x_def_pad);
  }
  {
    const int blocks = (WOo / TW) * (HOo / TH) * BB;    // 1024
    k_carafe<<<blocks, 256, 0, stream>>>(x_def_pad, w_sk, b_sk, w_km, b_km, out);
  }
}
